// WindowAttention_120259084937
// MI455X (gfx1250) — matrix operand order repaired and checked by reading, not compiled, not run on hardware
//
#include <hip/hip_runtime.h>

typedef __bf16 bf16_t;
typedef __attribute__((ext_vector_type(16))) __bf16 v16bf;
typedef __attribute__((ext_vector_type(8)))  float  v8f;
typedef __attribute__((ext_vector_type(4)))  int    i4;

#define TOK   49
#define HEADS 12
#define HDIM  32
#define EMB   384
#define NWIN  64
#define WAVES 4
#define PST   72   // P row stride (bf16 elems), 144B rows: 16B aligned, conflict-friendly
#define VST   72   // Vt row stride
#define QK_SCALE 0.17677669529663687f   // 32^-0.5

#define WMMA_BF16(A, B, C) \
  __builtin_amdgcn_wmma_f32_16x16x32_bf16(false, (A), false, (B), (short)0, (C), false, false)

__device__ __forceinline__ void load8(float* d, const float* s) {
  float4 a = *(const float4*)s;
  float4 c = *(const float4*)(s + 4);
  d[0]=a.x; d[1]=a.y; d[2]=a.z; d[3]=a.w;
  d[4]=c.x; d[5]=c.y; d[6]=c.z; d[7]=c.w;
}

// Max-reduce across each 16-lane row via v_permlane16_b32 (VALU, no LDS traffic).
// Nibble selectors encode lane ^ {1,2,4,8} within the 16-lane row.
__device__ __forceinline__ float rowmax16(float x) {
  union { float f; unsigned u; } a, b;
  a.f = x;
  b.u = __builtin_amdgcn_permlane16(a.u, a.u, 0x67452301u, 0xEFCDAB89u, false, false); // ^1
  a.f = fmaxf(a.f, b.f);
  b.u = __builtin_amdgcn_permlane16(a.u, a.u, 0x54761032u, 0xDCFE98BAu, false, false); // ^2
  a.f = fmaxf(a.f, b.f);
  b.u = __builtin_amdgcn_permlane16(a.u, a.u, 0x32107654u, 0xBA98FEDCu, false, false); // ^4
  a.f = fmaxf(a.f, b.f);
  b.u = __builtin_amdgcn_permlane16(a.u, a.u, 0xFEDCBA98u, 0x76543210u, false, false); // ^8
  a.f = fmaxf(a.f, b.f);
  return a.f;
}

// ---------------------------------------------------------------------------
// Pre-pass: combined bias + mask + column-padding table, fragment-ordered.
// BT[w][h][mt][rt][n16][nt] (float): per (g) the main kernel reads one float4.
// ---------------------------------------------------------------------------
__global__ __launch_bounds__(256)
void build_bias_table(const float* __restrict__ mask, const float* __restrict__ bias_table,
                      const int* __restrict__ rel_index, float* __restrict__ bt) {
  const int i   = blockIdx.x * 256 + threadIdx.x;   // 768*4096 = 3,145,728 total
  const int nt  =  i        & 3;
  const int n16 = (i >> 2)  & 15;
  const int rt  = (i >> 6)  & 15;
  const int mt  = (i >> 10) & 3;
  const int wh  =  i >> 12;                         // 0..767
  const int h   = wh % HEADS;
  const int w   = wh / HEADS;
  const int row = mt * 16 + rt;
  const int col = nt * 16 + n16;
  float val;
  if (col < TOK) {
    const int rc = row > TOK - 1 ? TOK - 1 : row;   // pad rows: finite garbage, never stored
    val = bias_table[rel_index[rc * TOK + col] * HEADS + h]
        + mask[(size_t)w * TOK * TOK + rc * TOK + col];
  } else {
    val = -1e30f;                                   // pad cols: exp underflows to exact 0
  }
  bt[i] = val;
}

// ---------------------------------------------------------------------------
// Main kernel: one wave per (window, head).
// ---------------------------------------------------------------------------
template <bool USE_BT>
__global__ __launch_bounds__(WAVES * 32)
void swin_attn_wmma(const float* __restrict__ q, const float* __restrict__ k,
                    const float* __restrict__ v, const float* __restrict__ mask,
                    const float* __restrict__ bias_table, const int* __restrict__ rel_index,
                    const float* __restrict__ bt,
                    float* __restrict__ out, int total_tasks) {
  __shared__ alignas(16) bf16_t sP [WAVES][64][PST];     // unnormalized probs (query, key)
  __shared__ alignas(16) bf16_t sVt[WAVES][HDIM][VST];   // V transposed (dim, key)

  const int wave = threadIdx.x >> 5;
  const int lane = threadIdx.x & 31;
  const int task = blockIdx.x * WAVES + wave;
  if (task >= total_tasks) return;                        // wave-uniform branch

  const int bwin = task / HEADS;
  const int h    = task - bwin * HEADS;
  const int w    = bwin & (NWIN - 1);                     // window-in-image index
  const int n16  = lane & 15;
  const int hh   = lane >> 4;                             // half-wave id

  const size_t base = (size_t)bwin * TOK * EMB + (size_t)h * HDIM;
  const float* qp = q + base;
  const float* kp = k + base;
  const float* vp = v + base;
  float*       op = out + base;
  const float* maskw = mask + (size_t)w * TOK * TOK;
  const float4* btw  = USE_BT
      ? (const float4*)bt + ((size_t)(w * HEADS + h) * 1024) + n16   // 1024 float4 per (w,h)
      : nullptr;

  // ---- K as WMMA B-fragments (4 tiles of K=32 x N=16keys), straight from global ----
  v16bf Kb[4];
  #pragma unroll
  for (int nt = 0; nt < 4; ++nt) {
    int key = nt * 16 + n16; if (key > TOK - 1) key = TOK - 1;   // clamp pad keys
    float f[16];
    const float* src = kp + (size_t)key * EMB + hh * 8;
    load8(f, src); load8(f + 8, src + 16);
    #pragma unroll
    for (int j = 0; j < 16; ++j) Kb[nt][j] = (bf16_t)f[j];
  }

  // ---- V -> LDS transposed (dim-major); zero pad keys 49..63 ----
  {
    float vr[HDIM];
    int r = lane;                                   // rows 0..31, all valid
    load8(vr, vp + (size_t)r * EMB);           load8(vr + 8,  vp + (size_t)r * EMB + 8);
    load8(vr + 16, vp + (size_t)r * EMB + 16); load8(vr + 24, vp + (size_t)r * EMB + 24);
    #pragma unroll
    for (int d = 0; d < HDIM; ++d) sVt[wave][d][r] = (bf16_t)vr[d];

    int r2 = lane + 32;                             // rows 32..63
    if (r2 <= TOK - 1) {
      load8(vr, vp + (size_t)r2 * EMB);           load8(vr + 8,  vp + (size_t)r2 * EMB + 8);
      load8(vr + 16, vp + (size_t)r2 * EMB + 16); load8(vr + 24, vp + (size_t)r2 * EMB + 24);
    } else {
      #pragma unroll
      for (int d = 0; d < HDIM; ++d) vr[d] = 0.0f;  // keep pad keys NaN-free
    }
    #pragma unroll
    for (int d = 0; d < HDIM; ++d) sVt[wave][d][r2] = (bf16_t)vr[d];
  }

  // ---- Phase 1: S = (Q*scale)K^T + bias(+mask), exp(S - rowmax) -> LDS (bf16) ----
  #pragma unroll 1
  for (int mt = 0; mt < 4; ++mt) {
    // Q A-fragment: lane m=n16 holds dims {hh*8..+7} and {16+hh*8..+7} of row mt*16+m
    v16bf Qa;
    {
      int row = mt * 16 + n16; if (row > TOK - 1) row = TOK - 1;  // clamp pad rows
      const float* qr = qp + (size_t)row * EMB;
      float f[16];
      load8(f,     qr + hh * 8);
      load8(f + 8, qr + 16 + hh * 8);
      #pragma unroll
      for (int j = 0; j < 16; ++j) Qa[j] = (bf16_t)(f[j] * QK_SCALE);
    }

    v8f S[4];
    #pragma unroll
    for (int nt = 0; nt < 4; ++nt) {
      v8f z = {};
      S[nt] = WMMA_BF16(Qa, Kb[nt], z);
    }

    if (USE_BT) {
      const float4* btm = btw + (size_t)mt * 256;
      #pragma unroll
      for (int g = 0; g < 8; ++g) {
        float4 bm = btm[(g + 8 * hh) * 16];
        S[0][g] += bm.x; S[1][g] += bm.y; S[2][g] += bm.z; S[3][g] += bm.w;
      }
    } else {
      #pragma unroll
      for (int g = 0; g < 8; ++g) {
        int rr = mt * 16 + g + 8 * hh;
        int rc = rr > TOK - 1 ? TOK - 1 : rr;
        #pragma unroll
        for (int nt = 0; nt < 4; ++nt) {
          int col = nt * 16 + n16;
          float s = S[nt][g];
          if (col < TOK) {
            int idx = rel_index[rc * TOK + col];
            s += bias_table[idx * HEADS + h] + maskw[rc * TOK + col];
          } else {
            s = -1e30f;
          }
          S[nt][g] = s;
        }
      }
    }

    // row max (in-lane over 4 regs + VALU permlane butterfly), then store exp.
    // Row sums are NOT reduced here: they come for free from a P x ones WMMA.
    #pragma unroll
    for (int g = 0; g < 8; ++g) {
      float m0 = fmaxf(fmaxf(S[0][g], S[1][g]), fmaxf(S[2][g], S[3][g]));
      m0 = rowmax16(m0);
      int rr = mt * 16 + g + 8 * hh;
      sP[wave][rr][ 0 + n16] = (bf16_t)__expf(S[0][g] - m0);
      sP[wave][rr][16 + n16] = (bf16_t)__expf(S[1][g] - m0);
      sP[wave][rr][32 + n16] = (bf16_t)__expf(S[2][g] - m0);
      sP[wave][rr][48 + n16] = (bf16_t)__expf(S[3][g] - m0);
    }
  }

  // ---- Phase 2: O = P @ V, rowsum = P @ ones; normalize at store ----
  v16bf Vb[2][2];
  #pragma unroll
  for (int kt = 0; kt < 2; ++kt)
    #pragma unroll
    for (int nt2 = 0; nt2 < 2; ++nt2) {
      union { i4 i[2]; v16bf v; } u;
      const i4* p = (const i4*)&sVt[wave][nt2 * 16 + n16][kt * 32 + hh * 16];
      u.i[0] = p[0]; u.i[1] = p[1];
      Vb[kt][nt2] = u.v;
    }

  v16bf ones;
  #pragma unroll
  for (int j = 0; j < 16; ++j) ones[j] = (bf16_t)1.0f;

  #pragma unroll 1
  for (int mt = 0; mt < 4; ++mt) {
    union { i4 i[2]; v16bf v; } Pa[2];
    const int prow = mt * 16 + n16;
    #pragma unroll
    for (int kt = 0; kt < 2; ++kt) {
      const bf16_t* pb = &sP[wave][prow][kt * 32 + hh * 16];
      Pa[kt].i[0] = *(const i4*)pb;
      Pa[kt].i[1] = *(const i4*)(pb + 8);
    }
    v8f Osum = {};
    Osum = WMMA_BF16(Pa[0].v, ones, Osum);       // every column = row sum of P
    Osum = WMMA_BF16(Pa[1].v, ones, Osum);
    v8f O0 = {}, O1 = {};
    O0 = WMMA_BF16(Pa[0].v, Vb[0][0], O0);
    O0 = WMMA_BF16(Pa[1].v, Vb[1][0], O0);
    O1 = WMMA_BF16(Pa[0].v, Vb[0][1], O1);
    O1 = WMMA_BF16(Pa[1].v, Vb[1][1], O1);
    #pragma unroll
    for (int g = 0; g < 8; ++g) {
      int rr = mt * 16 + g + 8 * hh;
      if (rr < TOK) {
        float rinv = 1.0f / Osum[g];
        op[(size_t)rr * EMB + n16]      = O0[g] * rinv;   // 64B per half-wave, coalesced
        op[(size_t)rr * EMB + 16 + n16] = O1[g] * rinv;
      }
    }
  }
}

extern "C" void kernel_launch(void* const* d_in, const int* in_sizes, int n_in,
                              void* d_out, int out_size, void* d_ws, size_t ws_size,
                              hipStream_t stream) {
  const float* q          = (const float*)d_in[0];
  const float* k          = (const float*)d_in[1];
  const float* v          = (const float*)d_in[2];
  const float* mask       = (const float*)d_in[3];
  const float* bias_table = (const float*)d_in[4];
  const int*   rel_index  = (const int*)  d_in[5];
  float* out = (float*)d_out;

  const int B_ = in_sizes[0] / (TOK * EMB);      // 4096
  const int total = B_ * HEADS;                  // 49152 wave-tasks
  const int blocks = (total + WAVES - 1) / WAVES;

  const size_t bt_elems = (size_t)NWIN * HEADS * 64 * 64;   // 3,145,728 floats (12.6 MB)
  if (ws_size >= bt_elems * sizeof(float)) {
    float* bt = (float*)d_ws;
    build_bias_table<<<(int)(bt_elems / 256), 256, 0, stream>>>(mask, bias_table, rel_index, bt);
    swin_attn_wmma<true><<<blocks, WAVES * 32, 0, stream>>>(
        q, k, v, mask, bias_table, rel_index, bt, out, total);
  } else {
    swin_attn_wmma<false><<<blocks, WAVES * 32, 0, stream>>>(
        q, k, v, mask, bias_table, rel_index, nullptr, out, total);
  }
}
